// HawkesMFDebias_44719199486346
// MI455X (gfx1250) — matrix operand order repaired and checked by reading, not compiled, not run on hardware
//
#include <hip/hip_runtime.h>
#include <hip/hip_bf16.h>

typedef __attribute__((ext_vector_type(16))) _Float16 v16h;
typedef __attribute__((ext_vector_type(8)))  _Float16 v8h;
typedef __attribute__((ext_vector_type(2)))  _Float16 v2h;
typedef __attribute__((ext_vector_type(8)))  float    v8f;

#define WAVES 4          // waves per block (128 threads) -> 44.8 KB LDS/block
#define CC 500
#define LL 200

#define LOG2E   1.4426950408889634f
#define LN2     0.6931471805599453f
#define BLOG2E  1.0098865286222743f   // 0.7 * log2(e)

// branch-free softplus on raw v_exp_f32/v_log_f32 (they compute 2^x / log2 x):
// softplus(x) = max(x,0) + ln(1 + exp(-|x|))
__device__ __forceinline__ float sp(float x) {
    float e = __builtin_amdgcn_exp2f(-LOG2E * __builtin_fabsf(x));
    float l = __builtin_amdgcn_logf(1.f + e) * LN2;
    return __builtin_fmaxf(x, 0.f) + l;
}

// exp(-0.7*(q-t)) masked by (t<q), branch-free
__device__ __forceinline__ float decay(float t, float q) {
    float e = __builtin_amdgcn_exp2f(BLOG2E * (t - q));
    return t < q ? e : 0.f;
}

// load 16 halves from two 16B-aligned 8-half chunks into a v16h fragment
__device__ __forceinline__ v16h ld2(const _Float16* p0, const _Float16* p1) {
    v8h lo = *(const v8h*)p0;
    v8h hi = *(const v8h*)p1;
    v16h r;
#pragma unroll
    for (int i = 0; i < 8; ++i) { r[i] = lo[i]; r[i + 8] = hi[i]; }
    return r;
}

// pre-swizzled B fragment: 512 halves per fragment, lane reads 32 contiguous bytes
__device__ __forceinline__ v16h ld_frag_g(const _Float16* base, int frag, int lane) {
    const _Float16* p = base + ((size_t)frag << 9) + (lane << 4);
    return ld2(p, p + 8);
}

// ---------------------------------------------------------------------------
// One-shot weight swizzle: f32 W1/W2 (base+exc) -> f16 B-fragment layout.
// B 32x16 f16 layout: N = nt*16 + (lane&15); K = ks*32 + (i&7) + 8*(lane>=16) + 16*(i>=8).
// w1s: [mlp][nt:8][ks:2][lane:32][i:16]   (16384 halves)
// w2s: [mlp][nt:8][ks:4][lane:32][i:16]   (32768 halves)
// ---------------------------------------------------------------------------
__global__ void prep_weights(const float* __restrict__ bW1, const float* __restrict__ eW1,
                             const float* __restrict__ bW2, const float* __restrict__ eW2,
                             _Float16* __restrict__ w1s, _Float16* __restrict__ w2s) {
    int gid = blockIdx.x * blockDim.x + threadIdx.x;
    if (gid < 16384) {
        int i    = gid & 15;
        int lane = (gid >> 4) & 31;
        int ks   = (gid >> 9) & 1;
        int nt   = (gid >> 10) & 7;
        int mlp  = gid >> 13;
        int K = ks * 32 + (i & 7) + ((lane >> 4) << 3) + ((i >> 3) << 4);     // < 64
        int N = nt * 16 + (lane & 15);                // < 128
        const float* W = mlp ? eW1 : bW1;
        w1s[gid] = (_Float16)W[K * 128 + N];
    } else {
        int g    = gid - 16384;                       // < 32768
        int i    = g & 15;
        int lane = (g >> 4) & 31;
        int ks   = (g >> 9) & 3;
        int nt   = (g >> 11) & 7;
        int mlp  = g >> 14;
        int K = ks * 32 + (i & 7) + ((lane >> 4) << 3) + ((i >> 3) << 4);     // < 128
        int N = nt * 16 + (lane & 15);
        const float* W = mlp ? eW2 : bW2;
        w2s[g] = (_Float16)W[K * 128 + N];
    }
}

// ---------------------------------------------------------------------------
// Fused kernel: per wave, one 16-row tile of the flattened (B*C) dimension.
//   Phase 1: Hawkes decay sums (streams 102 MB of histories, bandwidth-bound)
//   Phase 2: embedding gather -> f16 LDS tile (L2-resident table)
//   Phase 3: dual 64->128->128->1 MLP via v_wmma_f32_16x16x32_f16 (96 WMMAs)
// ---------------------------------------------------------------------------
__global__ __launch_bounds__(WAVES * 32)
void hawkes_mlp_kernel(const int*   __restrict__ items,
                       const float* __restrict__ qtimes,
                       const float* __restrict__ hist,
                       const float* __restrict__ emb,
                       const float* __restrict__ b_b1, const float* __restrict__ b_b2,
                       const float* __restrict__ b_W3,
                       const float* __restrict__ e_b1, const float* __restrict__ e_b2,
                       const float* __restrict__ e_W3,
                       const _Float16* __restrict__ w1s, const _Float16* __restrict__ w2s,
                       float* __restrict__ out) {
    // padded rows: 72-half (z) / 136-half (h1) strides => conflict-free ds_load_b128
    __shared__ alignas(16) _Float16 sh_z [WAVES][16][72];
    __shared__ alignas(16) _Float16 sh_h1[WAVES][2][16][136];
    __shared__ float sh_h[WAVES][16];
    __shared__ float sh_y[WAVES][2][16];

    const int wv   = threadIdx.x >> 5;
    const int lane = threadIdx.x & 31;
    const int tile = blockIdx.x * WAVES + wv;      // 8000 tiles exactly
    const int row0 = tile * 16;
    const int m    = lane & 15;
    const int chi  = (lane >> 4) << 3;             // 0 or 8

    // ---- Phase 1: h[r] = sum_L exp(-beta*(q-t)) * (t<q) -------------------
    {
        int b = row0 / CC;                          // uniform -> SALU
        int c = row0 % CC;
        for (int r = 0; r < 16; ++r) {
            const float  q  = qtimes[b];
            const float* hp = hist + (size_t)(row0 + r) * LL;
            float s = 0.f;
            // 200 floats = 32 lanes * float4  +  18 lanes * float4
            const float4 v0 = ((const float4*)hp)[lane];
            s += decay(v0.x, q); s += decay(v0.y, q);
            s += decay(v0.z, q); s += decay(v0.w, q);
            if (lane < 18) {
                const float4 v1 = *(const float4*)(hp + 128 + lane * 4);
                s += decay(v1.x, q); s += decay(v1.y, q);
                s += decay(v1.z, q); s += decay(v1.w, q);
            }
#pragma unroll
            for (int msk = 16; msk >= 1; msk >>= 1) s += __shfl_xor(s, msk);
            if (lane == 0) sh_h[wv][r] = s;
            if (++c == CC) { c = 0; ++b; }
        }
    }

    // ---- Phase 2: z tile gather (16 rows x 64 f32 -> f16) -----------------
    for (int r = 0; r < 16; ++r) {
        const int idx = items[row0 + r];
        const float2 v = ((const float2*)(emb + (size_t)idx * 64))[lane];
        v2h p; p[0] = (_Float16)v.x; p[1] = (_Float16)v.y;
        *(v2h*)&sh_z[wv][r][lane * 2] = p;
    }
    // same-wave LDS RAW below: compiler orders via s_wait_dscnt (no barrier)

    // ---- Layer 1: A fragments (16x32 f16 layout), reused for both MLPs ----
    const v16h a0 = ld2(&sh_z[wv][m][chi],      &sh_z[wv][m][chi + 16]);
    const v16h a1 = ld2(&sh_z[wv][m][32 + chi], &sh_z[wv][m][32 + chi + 16]);

#pragma unroll
    for (int nt = 0; nt < 8; ++nt) {
        const v16h bB0 = ld_frag_g(w1s, (0 * 8 + nt) * 2 + 0, lane);
        const v16h bB1 = ld_frag_g(w1s, (0 * 8 + nt) * 2 + 1, lane);
        const v16h bE0 = ld_frag_g(w1s, (1 * 8 + nt) * 2 + 0, lane);
        const v16h bE1 = ld_frag_g(w1s, (1 * 8 + nt) * 2 + 1, lane);
        v8f aB = {}; v8f aE = {};
        aB = __builtin_amdgcn_wmma_f32_16x16x32_f16(false, a0, false, bB0, (short)0, aB, false, false);
        aB = __builtin_amdgcn_wmma_f32_16x16x32_f16(false, a1, false, bB1, (short)0, aB, false, false);
        aE = __builtin_amdgcn_wmma_f32_16x16x32_f16(false, a0, false, bE0, (short)0, aE, false, false);
        aE = __builtin_amdgcn_wmma_f32_16x16x32_f16(false, a1, false, bE1, (short)0, aE, false, false);
        const int   N  = nt * 16 + m;
        const float bb = b_b1[N], be = e_b1[N];
#pragma unroll
        for (int r = 0; r < 8; ++r) {              // D layout: M = r + chi, N per lane
            sh_h1[wv][0][chi + r][N] = (_Float16)sp(aB[r] + bb);
            sh_h1[wv][1][chi + r][N] = (_Float16)sp(aE[r] + be);
        }
    }

    // ---- Layer 2 + fused layer 3 (H->1 dot) -------------------------------
#pragma unroll
    for (int mlp = 0; mlp < 2; ++mlp) {
        const float* bias2 = mlp ? e_b2 : b_b2;
        const float* w3    = mlp ? e_W3 : b_W3;
        float part[8] = {0, 0, 0, 0, 0, 0, 0, 0};
#pragma unroll
        for (int nt = 0; nt < 8; ++nt) {
            v8f acc = {};
#pragma unroll
            for (int ks = 0; ks < 4; ++ks) {
                const v16h a  = ld2(&sh_h1[wv][mlp][m][ks * 32 + chi],
                                    &sh_h1[wv][mlp][m][ks * 32 + chi + 16]);
                const v16h bf = ld_frag_g(w2s, (mlp * 8 + nt) * 4 + ks, lane);
                acc = __builtin_amdgcn_wmma_f32_16x16x32_f16(false, a, false, bf, (short)0, acc, false, false);
            }
            const int   N   = nt * 16 + m;
            const float b2v = bias2[N], w3v = w3[N];
#pragma unroll
            for (int r = 0; r < 8; ++r) part[r] += sp(acc[r] + b2v) * w3v;
        }
        // reduce over the 16 lanes of each half (rows 0-7 / 8-15)
#pragma unroll
        for (int msk = 1; msk <= 8; msk <<= 1)
#pragma unroll
            for (int r = 0; r < 8; ++r) part[r] += __shfl_xor(part[r], msk);
        if (m == 0)
#pragma unroll
            for (int r = 0; r < 8; ++r) sh_y[wv][mlp][chi + r] = part[r];
    }

    // ---- Combine: out = (sp(y_base)+eps) + (sp(y_exc)+eps) * h ------------
    if (lane < 16) {
        const float mu    = sp(sh_y[wv][0][lane]) + 1e-8f;
        const float alpha = sp(sh_y[wv][1][lane]) + 1e-8f;
        out[row0 + lane] = mu + alpha * sh_h[wv][lane];
    }
}

extern "C" void kernel_launch(void* const* d_in, const int* in_sizes, int n_in,
                              void* d_out, int out_size, void* d_ws, size_t ws_size,
                              hipStream_t stream) {
    (void)in_sizes; (void)n_in; (void)out_size; (void)ws_size;
    const int*   items = (const int*)  d_in[0];
    const float* qt    = (const float*)d_in[1];
    const float* hist  = (const float*)d_in[2];
    const float* emb   = (const float*)d_in[3];
    const float* bW1   = (const float*)d_in[4];
    const float* bb1   = (const float*)d_in[5];
    const float* bW2   = (const float*)d_in[6];
    const float* bb2   = (const float*)d_in[7];
    const float* bW3   = (const float*)d_in[8];
    const float* eW1   = (const float*)d_in[9];
    const float* eb1   = (const float*)d_in[10];
    const float* eW2   = (const float*)d_in[11];
    const float* eb2   = (const float*)d_in[12];
    const float* eW3   = (const float*)d_in[13];

    _Float16* w1s = (_Float16*)d_ws;                       // 32768 B
    _Float16* w2s = (_Float16*)((char*)d_ws + 32768);      // 65536 B
    float*    out = (float*)d_out;

    prep_weights<<<192, 256, 0, stream>>>(bW1, eW1, bW2, eW2, w1s, w2s);
    // 128000 rows / 16 = 8000 tiles; 4 waves/block -> 2000 blocks
    hawkes_mlp_kernel<<<2000, WAVES * 32, 0, stream>>>(
        items, qt, hist, emb, bb1, bb2, bW3, eb1, eb2, eW3, w1s, w2s, out);
}
